// SpikingSyntropyNN_33904471834878
// MI455X (gfx1250) — matrix operand order repaired and checked by reading, not compiled, not run on hardware
//
#include <hip/hip_runtime.h>
#include <hip/hip_bf16.h>
#include <stdint.h>

// ---------------------------------------------------------------------------
// Fused 3-layer LIF SNN for MI455X (gfx1250, wave32).
//  - f16 WMMA (16x16x32, f32 acc) for all three GEMMs; spikes are exact in f16.
//  - Each 256-thread WG (8 waves) owns a 64-row batch stripe for ALL layers and
//    ALL timesteps; spikes live in LDS, membrane states round-trip through L2.
//  - Weight chunks are staged into LDS by the Tensor Data Mover
//    (tensor_load_to_lds) with double buffering: wave 0 issues the DMA for
//    chunk i+1 (hardware row padding) while all waves run WMMA on chunk i;
//    completion tracked with TENSORcnt / s_wait_tensorcnt.
//  - Double-buffer selection is done with integer offsets from the `smem`
//    base (NOT a pointer array) so B-fragment loads keep LDS address-space
//    inference and lower to ds_load_b128 with immediate offsets.
// ---------------------------------------------------------------------------

typedef __attribute__((ext_vector_type(16))) _Float16 v16h;
typedef __attribute__((ext_vector_type(8)))  _Float16 v8h;
typedef __attribute__((ext_vector_type(8)))  float    v8f;
typedef __attribute__((ext_vector_type(4)))  unsigned int v4u;
typedef __attribute__((ext_vector_type(4)))  int v4i;
typedef __attribute__((ext_vector_type(8)))  int v8i;
typedef unsigned int u32;
typedef unsigned long long u64;

#define B_ROWS 16384
#define D_DIM  1024
#define H_DIM  512
#define MT     64      // batch rows per workgroup
#define KC     32      // WMMA K per step (16x16x32 f16)
#define SROW   520     // padded spike-row stride in halves (512 + 8)
#define XROW   40      // padded x-stage row stride (32 + 8)
#define WROW   40      // padded weight-stage row stride (32 + 8) == TDM padding
#define THREADS 256
#define WBUF_HALVES (H_DIM * WROW)                   // one weight buffer
#define SMEM_HALVES (MT * SROW + 2 * WBUF_HALVES)    // spikes + 2 weight buffers
#define SMEM_BYTES  (SMEM_HALVES * 2)                // 148,480 B (<320KB WGP LDS)

#define THR_V  1.0f
#define BETA_V 0.95f

#if __has_builtin(__builtin_amdgcn_tensor_load_to_lds)
#define HAVE_TDM 1
#else
#define HAVE_TDM 0
#endif

// ---------------------------------------------------------------------------
// TDM: DMA a [rows x 32] f16 tile (row stride ld_elems in memory) into LDS at
// lds_byte_addr, hardware-padding each 64B row by 16B -> 40-half LDS rows.
// D# packing per CDNA5 ISA 8.3/8.4. 2D tile: groups 2/3 zero.
// ---------------------------------------------------------------------------
#if HAVE_TDM
__device__ __forceinline__ void tdm_load_w(const _Float16* gsrc, u32 lds_byte_addr,
                                           int rows, int ld_elems) {
  u64 ga = (u64)(uintptr_t)gsrc;
  v4u g0 = {0, 0, 0, 0};
  g0[0] = 1u;                                        // count=1, user mode
  g0[1] = lds_byte_addr;                             // lds_addr [63:32]
  g0[2] = (u32)(ga & 0xffffffffu);                   // global_addr [95:64]
  g0[3] = (u32)((ga >> 32) & 0x01ffffffu)            // global_addr [120:96]
        | (2u << 30);                                // type=2 ("image") [127:126]
  u32 td0 = (u32)ld_elems;                           // tensor_dim0 (width, elems)
  u32 td1 = (u32)rows;                               // tensor_dim1 (rows)
  v8i g1 = {0, 0, 0, 0, 0, 0, 0, 0};
  g1[0] = (int)((1u << 16)                           // data_size = 2 bytes
              | (1u << 20)                           // pad_enable
              | (3u << 22)                           // pad_interval: 16 DWORDs (64B)
              | (3u << 25));                         // pad_amount:   4 DWORDs (16B)
  g1[1] = (int)((td0 & 0xffffu) << 16);              // tensor_dim0[15:0] @ [63:48]
  g1[2] = (int)(((td0 >> 16) & 0xffffu)              // tensor_dim0[31:16]
              | ((td1 & 0xffffu) << 16));            // tensor_dim1[15:0]
  g1[3] = (int)(((td1 >> 16) & 0xffffu)              // tensor_dim1[31:16]
              | ((u32)KC << 16));                    // tile_dim0 = 32 @ [127:112]
  g1[4] = (int)(td1 & 0xffffu);                      // tile_dim1 = rows; tile_dim2=0
  g1[5] = (int)td0;                                  // tensor_dim0_stride[31:0]
  v4i z4 = {0, 0, 0, 0};
#if defined(__clang_major__) && (__clang_major__ >= 23)
  v8i z8 = {0, 0, 0, 0, 0, 0, 0, 0};
  __builtin_amdgcn_tensor_load_to_lds(g0, g1, z4, z4, z8, 0);
#else
  __builtin_amdgcn_tensor_load_to_lds(g0, g1, z4, z4, 0);
#endif
}
#endif

// A fragment (16x32 f16 MxK) from row-major source, leading dim ldh (halves).
// ISA 7.12.2: lane m=L&15, hi=L>>4; elems 0..7 -> K=hi*8+e, 8..15 -> 16+hi*8+e.
__device__ __forceinline__ v16h load_a_frag(const _Float16* base, int ldh, int lane) {
  const int m = lane & 15, hi = lane >> 4;
  const _Float16* p = base + m * ldh + hi * 8;
  v8h lo = *(const v8h*)p;
  v8h hh = *(const v8h*)(p + 16);
  v16h a;
#pragma unroll
  for (int i = 0; i < 8; ++i) { a[i] = lo[i]; a[i + 8] = hh[i]; }
  return a;
}

// B fragment (32x16 f16 KxN) for out = S @ W^T with row-major W[n][k]:
// lane n=L&15, hi=L>>4 reads W[n0+n][k0 + hi*8 .. +7] and W[n0+n][k0 + 16 + hi*8 .. +7].
__device__ __forceinline__ v16h load_b_frag(const _Float16* base, int ldh, int lane) {
  const int n = lane & 15, hi = lane >> 4;
  const _Float16* p = base + n * ldh + hi * 8;
  v8h x0 = *(const v8h*)p;
  v8h x1 = *(const v8h*)(p + 16);
  v16h b;
#pragma unroll
  for (int i = 0; i < 8; ++i) { b[i] = x0[i]; b[i + 8] = x1[i]; }
  return b;
}

__device__ __forceinline__ float lif_update(float mprev, float cur, float& spk) {
  // snntorch Leaky, reset_mechanism='subtract' (reset from previous mem)
  float reset = (mprev > THR_V) ? 1.0f : 0.0f;
  float m = BETA_V * mprev + cur - reset * THR_V;
  spk = (m > THR_V) ? 1.0f : 0.0f;
  return m;
}

// ---------------------------------------------------------------------------
__global__ void convert_weights(const float* __restrict__ W1,
                                const float* __restrict__ W2,
                                const float* __restrict__ W3,
                                _Float16* __restrict__ W1h,
                                _Float16* __restrict__ W2h,
                                _Float16* __restrict__ W3h) {
  const int i0 = blockIdx.x * blockDim.x + threadIdx.x;
  const int stride = gridDim.x * blockDim.x;
  for (int i = i0; i < H_DIM * D_DIM; i += stride) W1h[i] = (_Float16)W1[i];
  for (int i = i0; i < H_DIM * H_DIM; i += stride) W2h[i] = (_Float16)W2[i];
  for (int i = i0; i < D_DIM * H_DIM; i += stride) W3h[i] = (_Float16)W3[i];
}

// ---------------------------------------------------------------------------
__global__ __launch_bounds__(THREADS, 1)
void snn_fused(const float* __restrict__ x,
               const float* __restrict__ b1,
               const float* __restrict__ b2,
               const float* __restrict__ b3,
               const _Float16* __restrict__ W1h,
               const _Float16* __restrict__ W2h,
               const _Float16* __restrict__ W3h,
               float* __restrict__ cur1,
               float* __restrict__ m1s,
               float* __restrict__ m2s,
               float* __restrict__ m3s,
               float* __restrict__ out,
               const int* __restrict__ nsteps) {
  extern __shared__ _Float16 smem[];
  _Float16* sbuf = smem;   // [MT][SROW] spikes / x-stage
  // Weight double buffers live at smem + MT*SROW + sel*WBUF_HALVES.
  // Always index via this offset arithmetic (rooted at `smem`) so loads stay
  // in addrspace(3) and lower to ds_load_b128.
#if HAVE_TDM
  const u32 wb_lds0 = (u32)(uintptr_t)(smem + MT * SROW);  // LDS byte address
#endif

  const int tid   = threadIdx.x;
  const int lane  = tid & 31;
  const int wave  = tid >> 5;
  const int band  = wave >> 1;   // 0..3 : rows band*16 .. +16 of the stripe
  const int cg    = wave & 1;    // column-group within N
  const int nlane = lane & 15;
  const int hi    = lane >> 4;
  const int r0    = blockIdx.x * MT;

  const int   T    = nsteps[0];
  const float invT = 1.0f / (float)T;

  // ---------------- Phase 0 (once): cur1 = x @ W1^T + b1 --------------------
  {
    v8f acc[16];
#pragma unroll
    for (int nt = 0; nt < 16; ++nt) {
      float bv = b1[cg * 256 + nt * 16 + nlane];
#pragma unroll
      for (int j = 0; j < 8; ++j) acc[nt][j] = bv;
    }
#if HAVE_TDM
    if (wave == 0) tdm_load_w(W1h, wb_lds0, H_DIM, D_DIM);     // prefetch chunk 0
#endif
    for (int ic = 0; ic < D_DIM / KC; ++ic) {
      const int kc = ic * KC;
      const int wboff = MT * SROW + (ic & 1) * WBUF_HALVES;
      __syncthreads();
      // stage x chunk [64 x 32] fp32 -> f16 (TDM cannot convert; VALU path)
      for (int i = tid; i < MT * KC; i += THREADS) {
        int r = i >> 5, k = i & 31;
        sbuf[r * XROW + k] = (_Float16)x[(size_t)(r0 + r) * D_DIM + kc + k];
      }
#if HAVE_TDM
      if (wave == 0) {
        if (ic + 1 < D_DIM / KC) {
          tdm_load_w(W1h + (kc + KC),
                     wb_lds0 + (u32)(((ic + 1) & 1) * WBUF_HALVES * 2), H_DIM, D_DIM);
          __builtin_amdgcn_s_wait_tensorcnt(1);   // chunk ic complete
        } else {
          __builtin_amdgcn_s_wait_tensorcnt(0);
        }
      }
#else
      for (int i = tid; i < H_DIM * 4; i += THREADS) {
        int n = i >> 2, kq = (i & 3) << 3;
        *(v8h*)&smem[wboff + n * WROW + kq] =
            *(const v8h*)&W1h[(size_t)n * D_DIM + kc + kq];
      }
#endif
      __syncthreads();
      v16h a = load_a_frag(&smem[(band * 16) * XROW], XROW, lane);
      v16h bcur = load_b_frag(&smem[wboff + (cg * 256) * WROW], WROW, lane);
#pragma unroll
      for (int nt = 0; nt < 16; ++nt) {           // pipelined: load nt+1, mma nt
        v16h bnext;
        if (nt < 15)
          bnext = load_b_frag(&smem[wboff + (cg * 256 + (nt + 1) * 16) * WROW], WROW, lane);
        acc[nt] = __builtin_amdgcn_wmma_f32_16x16x32_f16(
            false, a, false, bcur, (short)0, acc[nt], false, false);
        bcur = bnext;
      }
    }
#pragma unroll
    for (int nt = 0; nt < 16; ++nt) {
      int n = cg * 256 + nt * 16 + nlane;
#pragma unroll
      for (int j = 0; j < 8; ++j) {
        int row = r0 + band * 16 + hi * 8 + j;
        cur1[(size_t)row * H_DIM + n] = acc[nt][j];
      }
    }
    __threadfence();   // cur1 is read by other threads of this WG in phase 1
    __syncthreads();
  }

  // ---------------- Time loop -----------------------------------------------
  for (int t = 0; t < T; ++t) {
    const bool first = (t == 0);

    // ---- layer 1: elementwise LIF on cur1 -> s1 spikes into LDS
    for (int i = tid; i < MT * H_DIM; i += THREADS) {
      int r = i >> 9, c = i & (H_DIM - 1);
      size_t g = (size_t)(r0 + r) * H_DIM + c;
      float mp = first ? 0.0f : m1s[g];
      float spk;
      float m = lif_update(mp, cur1[g], spk);
      m1s[g] = m;
      sbuf[r * SROW + c] = (_Float16)spk;
    }
    __syncthreads();

    // ---- layer 2: out2 = s1 @ W2^T + b2 ; LIF -> s2 spikes into LDS
    {
      v8f acc[16];
#pragma unroll
      for (int nt = 0; nt < 16; ++nt) {
        float bv = b2[cg * 256 + nt * 16 + nlane];
#pragma unroll
        for (int j = 0; j < 8; ++j) acc[nt][j] = bv;
      }
#if HAVE_TDM
      if (wave == 0) tdm_load_w(W2h, wb_lds0, H_DIM, H_DIM);   // prefetch chunk 0
#endif
      for (int ic = 0; ic < H_DIM / KC; ++ic) {
        const int kc = ic * KC;
        const int wboff = MT * SROW + (ic & 1) * WBUF_HALVES;
        __syncthreads();   // buffer (ic+1)&1 free for refill
#if HAVE_TDM
        if (wave == 0) {
          if (ic + 1 < H_DIM / KC) {
            tdm_load_w(W2h + (kc + KC),
                       wb_lds0 + (u32)(((ic + 1) & 1) * WBUF_HALVES * 2), H_DIM, H_DIM);
            __builtin_amdgcn_s_wait_tensorcnt(1);
          } else {
            __builtin_amdgcn_s_wait_tensorcnt(0);
          }
        }
#else
        for (int i = tid; i < H_DIM * 4; i += THREADS) {
          int n = i >> 2, kq = (i & 3) << 3;
          *(v8h*)&smem[wboff + n * WROW + kq] =
              *(const v8h*)&W2h[(size_t)n * H_DIM + kc + kq];
        }
#endif
        __syncthreads();
        v16h a = load_a_frag(&smem[(band * 16) * SROW + kc], SROW, lane);
        v16h bcur = load_b_frag(&smem[wboff + (cg * 256) * WROW], WROW, lane);
#pragma unroll
        for (int nt = 0; nt < 16; ++nt) {
          v16h bnext;
          if (nt < 15)
            bnext = load_b_frag(&smem[wboff + (cg * 256 + (nt + 1) * 16) * WROW], WROW, lane);
          acc[nt] = __builtin_amdgcn_wmma_f32_16x16x32_f16(
              false, a, false, bcur, (short)0, acc[nt], false, false);
          bcur = bnext;
        }
      }
      __syncthreads();  // all waves done reading s1 -> safe to overwrite sbuf
#pragma unroll
      for (int nt = 0; nt < 16; ++nt) {
        int n = cg * 256 + nt * 16 + nlane;
#pragma unroll
        for (int j = 0; j < 8; ++j) {
          int rl = band * 16 + hi * 8 + j;
          size_t g = (size_t)(r0 + rl) * H_DIM + n;
          float mp = first ? 0.0f : m2s[g];
          float spk;
          float m = lif_update(mp, acc[nt][j], spk);
          m2s[g] = m;
          sbuf[rl * SROW + n] = (_Float16)spk;
        }
      }
      __syncthreads();
    }

    // ---- layer 3: out3 = s2 @ W3^T + b3 ; LIF -> accumulate spikes/T to out
    // 64 linearized chunks: c = nc*16 + ic ; 256-row W3 block nc, K-chunk ic.
    {
      v8f acc[8];
#if HAVE_TDM
      if (wave == 0) tdm_load_w(W3h, wb_lds0, 256, H_DIM);     // prefetch c = 0
#endif
      for (int c = 0; c < 64; ++c) {
        const int nc = c >> 4, ic = c & 15, kc = ic * KC;
        const int wboff = MT * SROW + (c & 1) * WBUF_HALVES;
        if (ic == 0) {
#pragma unroll
          for (int nt = 0; nt < 8; ++nt) {
            float bv = b3[nc * 256 + cg * 128 + nt * 16 + nlane];
#pragma unroll
            for (int j = 0; j < 8; ++j) acc[nt][j] = bv;
          }
        }
        __syncthreads();
#if HAVE_TDM
        if (wave == 0) {
          if (c + 1 < 64) {
            const int nn = (c + 1) >> 4, ii = (c + 1) & 15;
            tdm_load_w(W3h + (size_t)(nn * 256) * H_DIM + ii * KC,
                       wb_lds0 + (u32)(((c + 1) & 1) * WBUF_HALVES * 2), 256, H_DIM);
            __builtin_amdgcn_s_wait_tensorcnt(1);
          } else {
            __builtin_amdgcn_s_wait_tensorcnt(0);
          }
        }
#else
        for (int i = tid; i < 256 * 4; i += THREADS) {
          int n = i >> 2, kq = (i & 3) << 3;
          *(v8h*)&smem[wboff + n * WROW + kq] =
              *(const v8h*)&W3h[(size_t)(nc * 256 + n) * H_DIM + kc + kq];
        }
#endif
        __syncthreads();
        v16h a = load_a_frag(&smem[(band * 16) * SROW + kc], SROW, lane);
        v16h bcur = load_b_frag(&smem[wboff + (cg * 128) * WROW], WROW, lane);
#pragma unroll
        for (int nt = 0; nt < 8; ++nt) {
          v16h bnext;
          if (nt < 7)
            bnext = load_b_frag(&smem[wboff + (cg * 128 + (nt + 1) * 16) * WROW], WROW, lane);
          acc[nt] = __builtin_amdgcn_wmma_f32_16x16x32_f16(
              false, a, false, bcur, (short)0, acc[nt], false, false);
          bcur = bnext;
        }
        if (ic == 15) {   // finished K for this 256-col block: LIF + output
#pragma unroll
          for (int nt = 0; nt < 8; ++nt) {
            int n = nc * 256 + cg * 128 + nt * 16 + nlane;
#pragma unroll
            for (int j = 0; j < 8; ++j) {
              int row = r0 + band * 16 + hi * 8 + j;
              size_t g = (size_t)row * D_DIM + n;
              float mp = first ? 0.0f : m3s[g];
              float spk;
              float m = lif_update(mp, acc[nt][j], spk);
              m3s[g] = m;
              float contrib = spk * invT;
              out[g] = first ? contrib : (out[g] + contrib);  // t==0 overwrites poison
            }
          }
        }
      }
    }
    __syncthreads();
  }
}

// ---------------------------------------------------------------------------
extern "C" void kernel_launch(void* const* d_in, const int* in_sizes, int n_in,
                              void* d_out, int out_size, void* d_ws, size_t ws_size,
                              hipStream_t stream) {
  const float* x   = (const float*)d_in[0];
  const float* W1  = (const float*)d_in[1];
  const float* b1  = (const float*)d_in[2];
  const float* W2  = (const float*)d_in[3];
  const float* b2  = (const float*)d_in[4];
  const float* W3  = (const float*)d_in[5];
  const float* b3  = (const float*)d_in[6];
  const int* nstep = (const int*)d_in[7];
  float* out = (float*)d_out;

  // Workspace layout (~172 MB): f16 weights + cur1 + membrane states (L2-resident).
  char* ws = (char*)d_ws;
  size_t off = 0;
  _Float16* W1h = (_Float16*)(ws + off); off += (size_t)H_DIM * D_DIM * 2;
  _Float16* W2h = (_Float16*)(ws + off); off += (size_t)H_DIM * H_DIM * 2;
  _Float16* W3h = (_Float16*)(ws + off); off += (size_t)D_DIM * H_DIM * 2;
  float* cur1 = (float*)(ws + off); off += (size_t)B_ROWS * H_DIM * 4;
  float* m1s  = (float*)(ws + off); off += (size_t)B_ROWS * H_DIM * 4;
  float* m2s  = (float*)(ws + off); off += (size_t)B_ROWS * H_DIM * 4;
  float* m3s  = (float*)(ws + off); off += (size_t)B_ROWS * D_DIM * 4;

  // Allow >64KB dynamic LDS (WGP has 320KB; we use ~145KB). Idempotent.
  (void)hipFuncSetAttribute((const void*)snn_fused,
                            hipFuncAttributeMaxDynamicSharedMemorySize, SMEM_BYTES);

  convert_weights<<<1024, THREADS, 0, stream>>>(W1, W2, W3, W1h, W2h, W3h);
  snn_fused<<<B_ROWS / MT, THREADS, SMEM_BYTES, stream>>>(
      x, b1, b2, b3, W1h, W2h, W3h, cur1, m1s, m2s, m3s, out, nstep);
}
